// GCN_TCN_CapsNet_18279380812424
// MI455X (gfx1250) — matrix operand order repaired and checked by reading, not compiled, not run on hardware
//
#include <hip/hip_runtime.h>
#include <cstdint>

// ---------------------------------------------------------------------------
// GCN(knn) -> Conv1d(64->128,k5) -> BN -> ReLU -> Conv1d(128->256,k3) -> BN ->
// ReLU -> MaxPool(2) -> FC(16384->128) -> ReLU -> FC(128->6)
// B=2048, N=128.  Heavy GEMMs: v_wmma_f32_16x16x32_bf16 (wave32 CDNA5).
// Conv activation tiles staged into LDS via the Tensor Data Mover (TDM).
// ---------------------------------------------------------------------------

#define BATCH 2048
#define NPTS  128
#define BN_EPS 1e-5f

typedef __attribute__((ext_vector_type(16))) __bf16 v16b;
typedef __attribute__((ext_vector_type(8)))  __bf16 v8b;
typedef __attribute__((ext_vector_type(8)))  float  v8f;
typedef __attribute__((ext_vector_type(4)))  unsigned int u32x4;
typedef __attribute__((ext_vector_type(8)))  int i32x8;
typedef __attribute__((ext_vector_type(4)))  int i32x4;

__device__ __forceinline__ v8f wmma_bf16(v16b a, v16b b, v8f c) {
  return __builtin_amdgcn_wmma_f32_16x16x32_bf16(
      false, a, false, b, (short)0, c, false, false);
}

__device__ __forceinline__ v16b join8(v8b lo, v8b hi) {
  return __builtin_shufflevector(lo, hi, 0,1,2,3,4,5,6,7,8,9,10,11,12,13,14,15);
}

// A-matrix (16x32 bf16) per-lane K index (ISA 7.12.2): two contiguous 16B runs
__device__ __forceinline__ int a_kloc(int e, int hf) {
  int v = e >> 1;
  return 16 * (v >> 2) + 8 * hf + 2 * (v & 3) + (e & 1);
}
// B-matrix (32x16 bf16): per-lane K matches A: two 16B runs at 8*hf and 16+8*hf
__device__ __forceinline__ int b_kloc(int e, int hf) { return (e < 8) ? (8 * hf + e) : (16 + 8 * hf + (e - 8)); }

__device__ __forceinline__ v16b load_a_frag(const __bf16* row, int kb, int hf) {
  v8b lo = *(const v8b*)(row + kb + 8 * hf);
  v8b hi = *(const v8b*)(row + kb + 16 + 8 * hf);
  return join8(lo, hi);
}
__device__ __forceinline__ v16b load_b_frag(const __bf16* col, int kb, int hf) {
  const __bf16* p = col + kb + 8 * hf;
  return join8(*(const v8b*)p, *(const v8b*)(p + 16));
}

// ---------------------------------------------------------------------------
// Tensor Data Mover: 1-D DMA of nelem bf16 elements global -> LDS.
// D# per CDNA5 ISA ch.8: group0 = {count=1 | lds_addr | global_addr | type=2},
// group1 = {data_size=2B, tensor_dim0 = tile_dim0 = nelem, 1-D}.
// Issue from ONE wave; completion via TENSORcnt (s_wait_tensorcnt).
// 6-arg builtin form (clang-23 / therock-10.0 lane).
// ---------------------------------------------------------------------------
__device__ __forceinline__ void tdm_load_1d(const void* gsrc, void* ldst,
                                            unsigned nelem) {
  unsigned lds = (unsigned)(size_t)ldst;                 // LDS byte offset
  unsigned long long ga = (unsigned long long)(size_t)gsrc;
  u32x4 g0;
  g0[0] = 1u;                                            // count=1 (valid D#)
  g0[1] = lds;                                           // lds_addr
  g0[2] = (unsigned)(ga & 0xFFFFFFFFu);                  // global_addr[31:0]
  g0[3] = (unsigned)((ga >> 32) & 0x1FFFFFFu) | (2u << 30); // [56:32] | type=2
  i32x8 g1;
  g1[0] = 1 << 16;                                       // data_size=1 -> 2B
  g1[1] = (int)((nelem & 0xFFFFu) << 16);                // tensor_dim0[15:0]
  g1[2] = (int)((nelem >> 16) & 0xFFFFu);                // tensor_dim0[31:16]
  g1[3] = (int)((nelem & 0xFFFFu) << 16);                // tile_dim0
  g1[4] = 0;                                             // tile_dim1/2 = 0 (1-D)
  g1[5] = 0; g1[6] = 0; g1[7] = 0;                       // strides unused
  i32x4 z4 = {0, 0, 0, 0};
  i32x8 z8 = {0, 0, 0, 0, 0, 0, 0, 0};
  __builtin_amdgcn_tensor_load_to_lds(g0, g1, z4, z4, z8, 0);
}

// ===========================================================================
// Kernel A: per-sample KNN + normalized adjacency + 2 GCN layers (LDS WMMA).
// ===========================================================================
__global__ __launch_bounds__(256) void k_knn_gcn(
    const float* __restrict__ x,
    const float* __restrict__ W1, const float* __restrict__ b1,
    const float* __restrict__ W2, const float* __restrict__ b2,
    __bf16* __restrict__ y0h)
{
  __shared__ float  xs[NPTS][3];
  __shared__ short  nbr[NPTS][4];
  __shared__ float  dinv[NPTS];
  __shared__ __align__(16) __bf16 Adjh[NPTS][NPTS];  // 32 KB
  __shared__ __align__(16) __bf16 HBt[64][NPTS];     // 16 KB [feat][node]

  const int t    = threadIdx.x;
  const int bid  = blockIdx.x;
  const int lane = t & 31;
  const int wv   = t >> 5;
  const int hf   = lane >> 4;
  const int l15  = lane & 15;

  for (int idx = t; idx < NPTS * 3; idx += 256)
    xs[idx / 3][idx % 3] = x[(size_t)bid * NPTS * 3 + idx];
  __syncthreads();

  if (t < NPTS) {
    float x0 = xs[t][0], x1 = xs[t][1], x2 = xs[t][2];
    float bd0 = 3.4e38f, bd1 = 3.4e38f, bd2 = 3.4e38f, bd3 = 3.4e38f;
    int   bi0 = 0, bi1 = 0, bi2 = 0, bi3 = 0;
    for (int j = 0; j < NPTS; ++j) {
      if (j == t) continue;
      float d0 = x0 - xs[j][0], d1 = x1 - xs[j][1], d2 = x2 - xs[j][2];
      float d = d0 * d0 + d1 * d1 + d2 * d2;
      if (d < bd3) {
        if (d < bd0)      { bd3=bd2;bi3=bi2; bd2=bd1;bi2=bi1; bd1=bd0;bi1=bi0; bd0=d;bi0=j; }
        else if (d < bd1) { bd3=bd2;bi3=bi2; bd2=bd1;bi2=bi1; bd1=d;bi1=j; }
        else if (d < bd2) { bd3=bd2;bi3=bi2; bd2=d;bi2=j; }
        else              { bd3=d;bi3=j; }
      }
    }
    nbr[t][0] = (short)bi0; nbr[t][1] = (short)bi1;
    nbr[t][2] = (short)bi2; nbr[t][3] = (short)bi3;
  }
  __syncthreads();

  if (t < NPTS) {
    int cnt = 5;
    for (int j = 0; j < NPTS; ++j)
      cnt += (nbr[j][0]==t) + (nbr[j][1]==t) + (nbr[j][2]==t) + (nbr[j][3]==t);
    dinv[t] = rsqrtf((float)cnt);
  }
  __syncthreads();

  {
    int i  = t & 127;
    int c0 = (t >> 7) * 64;
    float di = dinv[i];
    short n0 = nbr[i][0], n1 = nbr[i][1], n2 = nbr[i][2], n3 = nbr[i][3];
    for (int j = c0; j < c0 + 64; ++j) {
      int m = (i == j) + (n0 == j) + (n1 == j) + (n2 == j) + (n3 == j)
            + (nbr[j][0] == i) + (nbr[j][1] == i) + (nbr[j][2] == i) + (nbr[j][3] == i);
      Adjh[i][j] = (__bf16)(m ? (float)m * di * dinv[j] : 0.0f);
    }
  }

  // Hlin = X @ W1 -> HBt[c][n]
  {
    int n = t >> 1;
    int cbase = (t & 1) * 16;
    float a0 = xs[n][0], a1 = xs[n][1], a2 = xs[n][2];
    #pragma unroll
    for (int c = 0; c < 16; ++c) {
      int cc = cbase + c;
      HBt[cc][n] = (__bf16)(a0 * W1[0*32+cc] + a1 * W1[1*32+cc] + a2 * W1[2*32+cc]);
    }
  }
  __syncthreads();

  // agg1 = relu(Adjh @ Hlin + b1)
  v8f acc1[2] = {{}, {}};
  {
    const __bf16* arow = &Adjh[(wv << 4) + l15][0];
    for (int ks = 0; ks < 4; ++ks) {
      v16b a = load_a_frag(arow, ks * 32, hf);
      #pragma unroll
      for (int nt = 0; nt < 2; ++nt) {
        v16b bb = load_b_frag(&HBt[nt * 16 + l15][0], ks * 32, hf);
        acc1[nt] = wmma_bf16(a, bb, acc1[nt]);
      }
    }
  }
  __syncthreads();
  #pragma unroll
  for (int nt = 0; nt < 2; ++nt)
    #pragma unroll
    for (int r = 0; r < 8; ++r) {
      int m = (wv << 4) + r + 8 * hf;
      int c = nt * 16 + l15;
      HBt[c][m] = (__bf16)fmaxf(acc1[nt][r] + b1[c], 0.0f);
    }
  __syncthreads();

  // H2lin = H1 @ W2 (single K-step)
  v8f acc2[4] = {{}, {}, {}, {}};
  {
    const int m = (wv << 4) + l15;
    v16b a;
    #pragma unroll
    for (int e = 0; e < 16; ++e) a[e] = HBt[a_kloc(e, hf)][m];
    #pragma unroll
    for (int nt = 0; nt < 4; ++nt) {
      v16b bb;
      #pragma unroll
      for (int e = 0; e < 16; ++e) bb[e] = (__bf16)W2[b_kloc(e, hf) * 64 + nt * 16 + l15];
      acc2[nt] = wmma_bf16(a, bb, acc2[nt]);
    }
  }
  __syncthreads();
  #pragma unroll
  for (int nt = 0; nt < 4; ++nt)
    #pragma unroll
    for (int r = 0; r < 8; ++r)
      HBt[nt * 16 + l15][(wv << 4) + r + 8 * hf] = (__bf16)acc2[nt][r];
  __syncthreads();

  // agg2 = relu(Adjh @ H2lin + b2)
  v8f acc3[4] = {{}, {}, {}, {}};
  {
    const __bf16* arow = &Adjh[(wv << 4) + l15][0];
    for (int ks = 0; ks < 4; ++ks) {
      v16b a = load_a_frag(arow, ks * 32, hf);
      #pragma unroll
      for (int nt = 0; nt < 4; ++nt) {
        v16b bb = load_b_frag(&HBt[nt * 16 + l15][0], ks * 32, hf);
        acc3[nt] = wmma_bf16(a, bb, acc3[nt]);
      }
    }
  }
  __syncthreads();
  #pragma unroll
  for (int nt = 0; nt < 4; ++nt)
    #pragma unroll
    for (int r = 0; r < 8; ++r) {
      int m = (wv << 4) + r + 8 * hf;
      int c = nt * 16 + l15;
      HBt[c][m] = (__bf16)fmaxf(acc3[nt][r] + b2[c], 0.0f);
    }
  __syncthreads();

  for (int i = 0; i < 32; ++i) {
    int idx = t + 256 * i;
    y0h[(size_t)bid * 8192 + idx] = (&HBt[0][0])[idx];
  }
}

// ===========================================================================
// Weight prep: fp32 -> bf16 swizzled into WMMA fragment order.
// ===========================================================================
__global__ __launch_bounds__(256) void k_prep_conv1(const float* __restrict__ W,
                                                    __bf16* __restrict__ Wf) {
  int t = blockIdx.x * 256 + threadIdx.x;
  if (t >= 8 * 10 * 32) return;
  int lane = t & 31, s = (t >> 5) % 10, oct = t / 320;
  int hf = lane >> 4, l15 = lane & 15;
  int oc = oct * 16 + l15;
  #pragma unroll
  for (int e = 0; e < 16; ++e) {
    int k = s * 32 + a_kloc(e, hf);
    Wf[(size_t)t * 16 + e] = (__bf16)W[(size_t)oc * 320 + k];
  }
}

__global__ __launch_bounds__(256) void k_prep_conv2(const float* __restrict__ W,
                                                    __bf16* __restrict__ Wf) {
  int t = blockIdx.x * 256 + threadIdx.x;
  if (t >= 16 * 12 * 32) return;
  int lane = t & 31, s = (t >> 5) % 12, oct = t / 384;
  int hf = lane >> 4, l15 = lane & 15;
  int oc = oct * 16 + l15;
  #pragma unroll
  for (int e = 0; e < 16; ++e) {
    int k = s * 32 + a_kloc(e, hf);
    Wf[(size_t)t * 16 + e] = (__bf16)W[(size_t)oc * 384 + k];
  }
}

__global__ __launch_bounds__(256) void k_prep_caps1(const float* __restrict__ W,
                                                    __bf16* __restrict__ Wf) {
  int t = blockIdx.x * 256 + threadIdx.x;
  if (t >= 8 * 512 * 32) return;
  int lane = t & 31, s = (t >> 5) & 511, ntile = t >> 14;
  int hf = lane >> 4, l15 = lane & 15;
  int n = ntile * 16 + l15;
  #pragma unroll
  for (int e = 0; e < 16; ++e) {
    int k = s * 32 + b_kloc(e, hf);
    Wf[(size_t)t * 16 + e] = (__bf16)W[(size_t)k * 128 + n];
  }
}

// ===========================================================================
// Conv1 (64->128, k=5, pad=2): TDM-stage y tile, build transposed patch,
// implicit GEMM W[128x320] @ Patch[320x128]; BN1 stats accumulated.
// ===========================================================================
__global__ __launch_bounds__(256) void k_conv1(
    const __bf16* __restrict__ y0h, const __bf16* __restrict__ W1f,
    const float* __restrict__ cb,
    float* __restrict__ gsum, float* __restrict__ gsq,
    __bf16* __restrict__ z1)
{
  __shared__ __align__(16) __bf16 stage[64 * 128];   // 16 KB raw y tile (TDM)
  __shared__ __align__(16) __bf16 P[128][320];       // 80 KB transposed patch
  __shared__ float ssum[128], ssq[128];

  const int t = threadIdx.x, bid = blockIdx.x;
  const int lane = t & 31, wv = t >> 5, hf = lane >> 4, l15 = lane & 15;

  if (wv == 0) {
    tdm_load_1d(y0h + (size_t)bid * 8192, stage, 8192);
    __builtin_amdgcn_s_wait_tensorcnt(0);
  }
  for (int i = t; i < 128; i += 256) { ssum[i] = 0.f; ssq[i] = 0.f; }
  __syncthreads();

  for (int i = 0; i < 160; ++i) {
    int idx = t + 256 * i;            // idx = p*128 + n
    int n = idx & 127, p = idx >> 7;
    int ic = p / 5, kk = p - 5 * ic;
    int sp = n + kk - 2;
    P[n][p] = (sp >= 0 && sp < 128) ? stage[ic * 128 + sp] : (__bf16)0.0f;
  }
  __syncthreads();

  v8f acc[8];
  #pragma unroll
  for (int nt = 0; nt < 8; ++nt) acc[nt] = {};

  for (int s = 0; s < 10; ++s) {
    v16b a = *(const v16b*)(W1f + ((size_t)(wv * 10 + s) * 32 + lane) * 16);
    #pragma unroll
    for (int nt = 0; nt < 8; ++nt) {
      v16b bb = load_b_frag(&P[nt * 16 + l15][0], s * 32, hf);
      acc[nt] = wmma_bf16(a, bb, acc[nt]);
    }
  }

  #pragma unroll
  for (int nt = 0; nt < 8; ++nt) {
    #pragma unroll
    for (int r = 0; r < 8; ++r) {
      int oc = (wv << 4) + r + 8 * hf;
      int n  = nt * 16 + l15;
      float v = acc[nt][r] + cb[oc];
      z1[((size_t)bid * 128 + oc) * 128 + n] = (__bf16)v;
      float sv = v, qv = v * v;
      #pragma unroll
      for (int m = 1; m < 16; m <<= 1) {
        sv += __shfl_xor(sv, m, 32);
        qv += __shfl_xor(qv, m, 32);
      }
      if (l15 == 0) { atomicAdd(&ssum[oc], sv); atomicAdd(&ssq[oc], qv); }
    }
  }
  __syncthreads();
  for (int i = t; i < 128; i += 256) {
    atomicAdd(&gsum[i], ssum[i]);
    atomicAdd(&gsq[i],  ssq[i]);
  }
}

// ===========================================================================
// BN finalize.
// ===========================================================================
__global__ void k_bnfin(const float* __restrict__ gsum, const float* __restrict__ gsq,
                        const float* __restrict__ g, const float* __restrict__ bta,
                        float* __restrict__ sc, float* __restrict__ sh, int nc)
{
  int t = blockIdx.x * blockDim.x + threadIdx.x;
  if (t < nc) {
    const float inv = 1.0f / ((float)BATCH * (float)NPTS);
    float m   = gsum[t] * inv;
    float var = gsq[t] * inv - m * m;
    float s   = g[t] * rsqrtf(var + BN_EPS);
    sc[t] = s;
    sh[t] = bta[t] - m * s;
  }
}

// ===========================================================================
// Conv2 (128->256, k=3, pad=1): TDM-stage z1 tile, BN1+ReLU fused into patch
// build, implicit GEMM W[256x384] @ Patch[384x128]; BN2 stats accumulated.
// ===========================================================================
__global__ __launch_bounds__(256) void k_conv2(
    const __bf16* __restrict__ z1,
    const float* __restrict__ s1, const float* __restrict__ t1,
    const __bf16* __restrict__ W2f, const float* __restrict__ cb,
    float* __restrict__ gsum, float* __restrict__ gsq,
    __bf16* __restrict__ z2)
{
  __shared__ __align__(16) __bf16 stage[128 * 128];  // 32 KB raw z1 tile (TDM)
  __shared__ __align__(16) __bf16 P[128][384];       // 96 KB transposed patch
  __shared__ float ssum[256], ssq[256];

  const int t = threadIdx.x, bid = blockIdx.x;
  const int lane = t & 31, wv = t >> 5, hf = lane >> 4, l15 = lane & 15;

  if (wv == 0) {
    tdm_load_1d(z1 + (size_t)bid * 16384, stage, 16384);
    __builtin_amdgcn_s_wait_tensorcnt(0);
  }
  for (int i = t; i < 256; i += 256) { ssum[i] = 0.f; ssq[i] = 0.f; }
  __syncthreads();

  for (int i = 0; i < 192; ++i) {
    int idx = t + 256 * i;            // idx = p*128 + n
    int n = idx & 127, p = idx >> 7;
    int ic = p / 3, kk = p - 3 * ic;
    int sp = n + kk - 1;
    float v = 0.0f;
    if (sp >= 0 && sp < 128)
      v = fmaxf((float)stage[ic * 128 + sp] * s1[ic] + t1[ic], 0.0f);
    P[n][p] = (__bf16)v;
  }
  __syncthreads();

  v8f acc[16];
  #pragma unroll
  for (int i = 0; i < 16; ++i) acc[i] = {};

  for (int s = 0; s < 12; ++s) {
    v16b a0 = *(const v16b*)(W2f + ((size_t)((2 * wv)     * 12 + s) * 32 + lane) * 16);
    v16b a1 = *(const v16b*)(W2f + ((size_t)((2 * wv + 1) * 12 + s) * 32 + lane) * 16);
    #pragma unroll
    for (int nt = 0; nt < 8; ++nt) {
      v16b bb = load_b_frag(&P[nt * 16 + l15][0], s * 32, hf);
      acc[nt]     = wmma_bf16(a0, bb, acc[nt]);
      acc[8 + nt] = wmma_bf16(a1, bb, acc[8 + nt]);
    }
  }

  #pragma unroll
  for (int oi = 0; oi < 2; ++oi) {
    #pragma unroll
    for (int nt = 0; nt < 8; ++nt) {
      #pragma unroll
      for (int r = 0; r < 8; ++r) {
        int oc = (2 * wv + oi) * 16 + r + 8 * hf;
        int n  = nt * 16 + l15;
        float v = acc[oi * 8 + nt][r] + cb[oc];
        z2[((size_t)bid * 256 + oc) * 128 + n] = (__bf16)v;
        float sv = v, qv = v * v;
        #pragma unroll
        for (int m = 1; m < 16; m <<= 1) {
          sv += __shfl_xor(sv, m, 32);
          qv += __shfl_xor(qv, m, 32);
        }
        if (l15 == 0) { atomicAdd(&ssum[oc], sv); atomicAdd(&ssq[oc], qv); }
      }
    }
  }
  __syncthreads();
  for (int i = t; i < 256; i += 256) {
    atomicAdd(&gsum[i], ssum[i]);
    atomicAdd(&gsq[i],  ssq[i]);
  }
}

// ===========================================================================
// BN2 + ReLU + MaxPool1d(2) -> Q[b][oc*64+p] bf16
// ===========================================================================
__global__ __launch_bounds__(256) void k_pool(
    const __bf16* __restrict__ z2,
    const float* __restrict__ s2, const float* __restrict__ t2,
    __bf16* __restrict__ Q)
{
  size_t o = (size_t)blockIdx.x * 256 + threadIdx.x;
  int p  = (int)(o & 63);
  int oc = (int)((o >> 6) & 255);
  size_t b = o >> 14;
  size_t base = (b * 256 + oc) * 128 + 2 * p;
  float sc = s2[oc], sh = t2[oc];
  float a = fmaxf((float)z2[base]     * sc + sh, 0.0f);
  float c = fmaxf((float)z2[base + 1] * sc + sh, 0.0f);
  Q[o] = (__bf16)fmaxf(a, c);
}

// ===========================================================================
// Swizzle pooled activations into A-fragment order for FC1.
// ===========================================================================
__global__ __launch_bounds__(256) void k_swz_q(const __bf16* __restrict__ Q,
                                               __bf16* __restrict__ Qf) {
  int t = blockIdx.x * 256 + threadIdx.x;
  if (t >= 128 * 512 * 32) return;
  int lane = t & 31, s = (t >> 5) & 511, mt = t >> 14;
  int hf = lane >> 4, l15 = lane & 15;
  const __bf16* q = Q + ((size_t)mt * 16 + l15) * 16384 + s * 32;
  #pragma unroll
  for (int e = 0; e < 16; ++e)
    Qf[(size_t)t * 16 + e] = q[a_kloc(e, hf)];
}

// ===========================================================================
// FC1: per K-step one 32B A load + one 32B B load + one WMMA.
// ===========================================================================
__global__ __launch_bounds__(256) void k_fc1(
    const __bf16* __restrict__ Qf, const __bf16* __restrict__ Cf,
    const float* __restrict__ bias, float* __restrict__ out1)
{
  const int t = threadIdx.x, mt = blockIdx.x;
  const int lane = t & 31, wv = t >> 5, hf = lane >> 4, l15 = lane & 15;

  v8f acc = {};
  for (int s = 0; s < 512; ++s) {
    v16b a  = *(const v16b*)(Qf + ((size_t)(mt * 512 + s) * 32 + lane) * 16);
    v16b bb = *(const v16b*)(Cf + ((size_t)(wv * 512 + s) * 32 + lane) * 16);
    acc = wmma_bf16(a, bb, acc);
  }
  #pragma unroll
  for (int r = 0; r < 8; ++r) {
    int m = mt * 16 + r + 8 * hf;
    int j = wv * 16 + l15;
    out1[(size_t)m * 128 + j] = fmaxf(acc[r] + bias[j], 0.0f);
  }
}

// ===========================================================================
// FC2: tiny f32 GEMV.
// ===========================================================================
__global__ __launch_bounds__(256) void k_fc2(
    const float* __restrict__ h, const float* __restrict__ W,
    const float* __restrict__ bias, float* __restrict__ out)
{
  int o = blockIdx.x * 256 + threadIdx.x;
  if (o >= BATCH * 6) return;
  int b = o / 6, c = o - b * 6;
  float acc = bias[c];
  const float* hr = h + (size_t)b * 128;
  for (int j = 0; j < 128; ++j) acc += hr[j] * W[j * 6 + c];
  out[o] = acc;
}

// ===========================================================================
extern "C" void kernel_launch(void* const* d_in, const int* in_sizes, int n_in,
                              void* d_out, int out_size, void* d_ws, size_t ws_size,
                              hipStream_t stream)
{
  const float* x    = (const float*)d_in[0];
  const float* g1W  = (const float*)d_in[1];
  const float* g1b  = (const float*)d_in[2];
  const float* g2W  = (const float*)d_in[3];
  const float* g2b  = (const float*)d_in[4];
  const float* c1W  = (const float*)d_in[5];
  const float* c1b  = (const float*)d_in[6];
  const float* bn1g = (const float*)d_in[7];
  const float* bn1b = (const float*)d_in[8];
  const float* c2W  = (const float*)d_in[9];
  const float* c2b  = (const float*)d_in[10];
  const float* bn2g = (const float*)d_in[11];
  const float* bn2b = (const float*)d_in[12];
  const float* cp1W = (const float*)d_in[13];
  const float* cp1b = (const float*)d_in[14];
  const float* cp2W = (const float*)d_in[15];
  const float* cp2b = (const float*)d_in[16];
  float* out = (float*)d_out;

  char* p = (char*)d_ws;
  auto take = [&](size_t bytes) -> char* {
    char* r = p;
    p += (bytes + 255) & ~(size_t)255;
    return r;
  };
  __bf16* y0h = (__bf16*)take((size_t)BATCH * 64  * 128 * 2);
  __bf16* z1  = (__bf16*)take((size_t)BATCH * 128 * 128 * 2);
  __bf16* z2  = (__bf16*)take((size_t)BATCH * 256 * 128 * 2);
  __bf16* Qh  = (__bf16*)take((size_t)BATCH * 16384 * 2);
  __bf16* Qf  = (__bf16*)take((size_t)BATCH * 16384 * 2);
  float*  h1  = (float*) take((size_t)BATCH * 128 * 4);
  __bf16* W1f = (__bf16*)take((size_t)128 * 320 * 2);
  __bf16* W2f = (__bf16*)take((size_t)256 * 384 * 2);
  __bf16* C1f = (__bf16*)take((size_t)16384 * 128 * 2);
  float* stats = (float*)take((size_t)(768 + 768) * 4);
  float* gsum1 = stats;          float* gsq1 = stats + 128;
  float* gsum2 = stats + 256;    float* gsq2 = stats + 512;
  float* s1 = stats + 768;       float* t1 = stats + 896;
  float* s2 = stats + 1024;      float* t2 = stats + 1280;

  (void)hipMemsetAsync(stats, 0, 768 * sizeof(float), stream);

  k_prep_conv1<<<10,  256, 0, stream>>>(c1W,  W1f);
  k_prep_conv2<<<24,  256, 0, stream>>>(c2W,  W2f);
  k_prep_caps1<<<512, 256, 0, stream>>>(cp1W, C1f);

  k_knn_gcn<<<BATCH, 256, 0, stream>>>(x, g1W, g1b, g2W, g2b, y0h);

  k_conv1<<<BATCH, 256, 0, stream>>>(y0h, W1f, c1b, gsum1, gsq1, z1);
  k_bnfin<<<1, 256, 0, stream>>>(gsum1, gsq1, bn1g, bn1b, s1, t1, 128);

  k_conv2<<<BATCH, 256, 0, stream>>>(z1, s1, t1, W2f, c2b, gsum2, gsq2, z2);
  k_bnfin<<<1, 256, 0, stream>>>(gsum2, gsq2, bn2g, bn2b, s2, t2, 256);

  k_pool<<<(BATCH * 256 * 64) / 256, 256, 0, stream>>>(z2, s2, t2, Qh);
  k_swz_q<<<(128 * 512 * 32) / 256, 256, 0, stream>>>(Qh, Qf);

  k_fc1<<<BATCH / 16, 256, 0, stream>>>(Qf, C1f, cp1b, h1);
  k_fc2<<<(BATCH * 6 + 255) / 256, 256, 0, stream>>>(h1, cp2W, cp2b, out);

  (void)in_sizes; (void)n_in; (void)out_size; (void)ws_size;
}
